// GraphAttentionLayer_69870527971509
// MI455X (gfx1250) — matrix operand order repaired and checked by reading, not compiled, not run on hardware
//
#include <hip/hip_runtime.h>
#include <hip/hip_bf16.h>
#include <math.h>

#define N_NODES 2048
#define IN_F    1024
#define N_HEADS 8
#define N_HID   32
#define OUT_COLS (N_HEADS * N_HID)   // 256
#define NEG_SLOPE 0.2f

typedef __attribute__((ext_vector_type(16))) __bf16       v16bf;
typedef __attribute__((ext_vector_type(8)))  float        v8f;
typedef __attribute__((ext_vector_type(4)))  float        v4f;
typedef __attribute__((ext_vector_type(4)))  unsigned int v4u;
typedef __attribute__((ext_vector_type(4)))  unsigned int u32x4;
typedef __attribute__((ext_vector_type(8)))  int          i32x8;
typedef __attribute__((ext_vector_type(4)))  int          i32x4;

union BF16x16 {
    v16bf v;
    __bf16 e[16];
    v4u q[2];
};

__device__ __forceinline__ float leaky(float x) { return x >= 0.f ? x : NEG_SLOPE * x; }

// ---------------------------------------------------------------------------
// TDM: async-load a 16x32 int32 tile of adj into LDS (D# per ISA ch.8).
//   group0: [1:0]count=1 | [63:32]lds_addr | [120:64]global_addr | [127:126]type=2
//   group1: data_size=2(4B), tensor_dim0/1=2048, tile_dim0=32, tile_dim1=16,
//           tensor_dim0_stride=2048
// ---------------------------------------------------------------------------
__device__ __forceinline__ void tdm_load_adj_tile(const int* adj, int i0, int jc,
                                                  unsigned lds_off)
{
    unsigned long long ga = (unsigned long long)(const void*)adj
                          + ((unsigned long long)i0 * N_NODES + (unsigned)jc) * 4ull;
    u32x4 g0;
    g0[0] = 1u;                                       // count=1, gather off
    g0[1] = lds_off;                                  // LDS byte address
    g0[2] = (unsigned)(ga & 0xFFFFFFFFu);             // global_addr[31:0]
    g0[3] = (unsigned)((ga >> 32) & 0x01FFFFFFu)      // global_addr[56:32]
          | (2u << 30);                               // type=2 ("image")
    i32x8 g1;
    g1[0] = (int)(2u << 16);       // data_size=2 -> 4 bytes; no pad/iterate/mask
    g1[1] = (int)0x08000000;       // tensor_dim0 = 2048 (low16 at bits[63:48])
    g1[2] = (int)0x08000000;       // tensor_dim0 hi=0; tensor_dim1=2048 low16
    g1[3] = (int)(32u << 16);      // tensor_dim1 hi=0; tile_dim0 = 32
    g1[4] = 16;                    // tile_dim1 = 16; tile_dim2 = 0
    g1[5] = 2048;                  // tensor_dim0_stride = 2048 (low32)
    g1[6] = 0;                     // stride0 hi16; tensor_dim1_stride low16
    g1[7] = 0;
    i32x4 gz = {};
#if __clang_major__ >= 23
    i32x8 gz8 = {};
    __builtin_amdgcn_tensor_load_to_lds(g0, g1, gz, gz, gz8, 0);
#else
    __builtin_amdgcn_tensor_load_to_lds(g0, g1, gz, gz, 0);
#endif
}

// ---------------------------------------------------------------------------
// Kernel 1: g = h @ W  (2048x1024 @ 1024x256), stored transposed as bf16:
//           gT[col][node], col = head*32 + f.  One wave per 16x16 tile,
//           K in chunks of 32 via v_wmma_f32_16x16x32_bf16.
// ---------------------------------------------------------------------------
__global__ void __launch_bounds__(256) k_gemm_g(const float* __restrict__ h,
                                                const float* __restrict__ W,
                                                __bf16* __restrict__ gT)
{
    const int wave = threadIdx.x >> 5;
    const int lane = threadIdx.x & 31;
    const int tile = blockIdx.x * 8 + wave;      // 0..2047
    const int it = tile >> 4;                    // 0..127  (i tile)
    const int nt = tile & 15;                    // 0..15   (col tile)
    const int i0 = it * 16, n0 = nt * 16;
    const int rloc = lane & 15;
    const int half = lane >> 4;

    v8f c = {};
    const float* hrow = h + (size_t)(i0 + rloc) * IN_F;
    const int ncol = n0 + rloc;

    for (int kc = 0; kc < IN_F; kc += 32) {
        // ---- A fragment (h tile, fp32 -> bf16), k = kc + 16*half + e (contiguous)
        BF16x16 A;
        const int ks1 = kc + half * 16;
        const int ks2 = kc + half * 16 + 8;
        v4f a0 = *(const v4f*)(hrow + ks1);
        v4f a1 = *(const v4f*)(hrow + ks1 + 4);
        v4f a2 = *(const v4f*)(hrow + ks2);
        v4f a3 = *(const v4f*)(hrow + ks2 + 4);
        A.e[0]  = (__bf16)a0.x; A.e[1]  = (__bf16)a0.y;
        A.e[2]  = (__bf16)a0.z; A.e[3]  = (__bf16)a0.w;
        A.e[4]  = (__bf16)a1.x; A.e[5]  = (__bf16)a1.y;
        A.e[6]  = (__bf16)a1.z; A.e[7]  = (__bf16)a1.w;
        A.e[8]  = (__bf16)a2.x; A.e[9]  = (__bf16)a2.y;
        A.e[10] = (__bf16)a2.z; A.e[11] = (__bf16)a2.w;
        A.e[12] = (__bf16)a3.x; A.e[13] = (__bf16)a3.y;
        A.e[14] = (__bf16)a3.z; A.e[15] = (__bf16)a3.w;

        // ---- B fragment (W tile): lane = col, 16 K values (coalesced across lanes)
        BF16x16 B;
        const int kb = kc + half * 16;
#pragma unroll
        for (int t = 0; t < 16; ++t)
            B.e[t] = (__bf16)W[(size_t)(kb + t) * OUT_COLS + ncol];

        c = __builtin_amdgcn_wmma_f32_16x16x32_bf16(false, A.v, false, B.v,
                                                    (short)0, c, false, false);
    }

    // store transposed (gT[col][node]) as bf16; consecutive r -> contiguous bytes
#pragma unroll
    for (int r = 0; r < 8; ++r) {
        const int M = r + half * 8;
        gT[(size_t)ncol * N_NODES + (i0 + M)] = (__bf16)c[r];
    }
}

// ---------------------------------------------------------------------------
// Kernel 2: s_l[i,h] = g[i,h,:]·a_l ; s_rT[h,j] = g[j,h,:]·a_r
// ---------------------------------------------------------------------------
__global__ void __launch_bounds__(256) k_scores(const __bf16* __restrict__ gT,
                                                const float* __restrict__ a,
                                                float* __restrict__ s_l,
                                                float* __restrict__ s_rT)
{
    const int t = blockIdx.x * 256 + threadIdx.x;   // 0..16383
    const int i = t >> 3;
    const int hh = t & 7;
    float accl = 0.f, accr = 0.f;
    const __bf16* base = gT + (size_t)hh * N_HID * N_NODES + i;
#pragma unroll
    for (int f = 0; f < N_HID; ++f) {
        const float g = (float)base[(size_t)f * N_NODES];
        accl += g * a[f];
        accr += g * a[N_HID + f];
    }
    s_l[i * N_HEADS + hh]  = accl;
    s_rT[hh * N_NODES + i] = accr;
}

// ---------------------------------------------------------------------------
// Kernel 3: per (i,h) row: masked online softmax stats -> m, 1/Z
// ---------------------------------------------------------------------------
__global__ void __launch_bounds__(256) k_stats(const int* __restrict__ adj,
                                               const float* __restrict__ s_l,
                                               const float* __restrict__ s_rT,
                                               float* __restrict__ mrow,
                                               float* __restrict__ invZ)
{
    const int wave = threadIdx.x >> 5;
    const int lane = threadIdx.x & 31;
    const int pair = blockIdx.x * 8 + wave;   // 0..16383
    const int i = pair >> 3;
    const int hh = pair & 7;
    const float sl = s_l[i * N_HEADS + hh];
    const int* arow = adj + (size_t)i * N_NODES;
    const float* sr = s_rT + (size_t)hh * N_NODES;

    float m = -__builtin_inff(), s = 0.f;
    for (int j = lane; j < N_NODES; j += 32) {
        if (arow[j] != 0) {
            const float e = leaky(sl + sr[j]);
            const float mn = fmaxf(m, e);
            s = s * __expf(m - mn) + __expf(e - mn);
            m = mn;
        }
    }
#pragma unroll
    for (int off = 16; off >= 1; off >>= 1) {
        const float m2 = __shfl_xor(m, off, 32);
        const float s2 = __shfl_xor(s, off, 32);
        const float mn = fmaxf(m, m2);
        float sn = 0.f;
        if (mn > -__builtin_inff())
            sn = s * __expf(m - mn) + s2 * __expf(m2 - mn);
        m = mn; s = sn;
    }
    if (lane == 0) {
        mrow[i * N_HEADS + hh] = m;
        invZ[i * N_HEADS + hh] = (s > 0.f) ? 1.f / s : 0.f;
    }
}

// ---------------------------------------------------------------------------
// Kernel 4: fused attention: out[i,h,:] = sum_j p_ij * g[j,h,:]
// Block = one 16-row i-tile; wave w = head w.  Adjacency tiles are DMA'd
// into a double-buffered LDS region by the Tensor Data Mover (TENSORcnt),
// overlapping with the WMMA consume of the previous tile.
// Dynamic LDS layout: [0,2048)=adj buf0, [2048,4096)=adj buf1, [4096,...)=s_rT
// ---------------------------------------------------------------------------
__global__ void __launch_bounds__(256) k_attn(const int* __restrict__ adj,
                                              const __bf16* __restrict__ gT,
                                              const float* __restrict__ s_l,
                                              const float* __restrict__ s_rT,
                                              const float* __restrict__ mrow,
                                              const float* __restrict__ invZ,
                                              float* __restrict__ out)
{
    extern __shared__ char smem[];
    float* sh_sr = (float*)(smem + 4096);

    const int tid  = threadIdx.x;
    const int hh   = tid >> 5;                   // wave == head
    const int lane = tid & 31;
    const int i0   = blockIdx.x * 16;
    const int rloc = lane & 15;
    const int half = lane >> 4;

    // wave0 kicks off the first adjacency tile DMA, overlapped with s_rT staging
    if (hh == 0)
        tdm_load_adj_tile(adj, i0, 0, 0u);

    for (int t = tid; t < N_HEADS * N_NODES; t += 256) sh_sr[t] = s_rT[t];

    const int irow = i0 + rloc;
    const float sl = s_l[irow * N_HEADS + hh];
    const float mr = mrow[irow * N_HEADS + hh];
    const float iz = invZ[irow * N_HEADS + hh];
    const float* srh = sh_sr + hh * N_NODES;

    v8f c0 = {}, c1 = {};
    const __bf16* gb0 = gT + (size_t)(hh * N_HID + rloc) * N_NODES;        // f = lane%16
    const __bf16* gb1 = gT + (size_t)(hh * N_HID + 16 + rloc) * N_NODES;   // f = 16+lane%16

    for (int jc = 0; jc < N_NODES; jc += 32) {
        const int k = jc >> 5;
        if (hh == 0)
            __builtin_amdgcn_s_wait_tensorcnt((short)0);   // tile k resident in LDS
        __syncthreads();                                    // publish tile k; tile k-1 reads done
        if (hh == 0 && jc + 32 < N_NODES)
            tdm_load_adj_tile(adj, i0, jc + 32, (unsigned)(((k + 1) & 1) * 2048));

        const int* sh_adj = (const int*)(smem + (k & 1) * 2048);

        // ---- A fragment: p = exp(leaky(sl+sr) - m) * invZ  (0 where masked)
        BF16x16 A;
#pragma unroll
        for (int e = 0; e < 16; ++e) {
            const int K = half * 16 + e;
            float p = 0.f;
            if (sh_adj[rloc * 32 + K] != 0) {
                const float ev = leaky(sl + srh[jc + K]);
                p = __expf(ev - mr) * iz;
            }
            A.e[e] = (__bf16)p;
        }

        // ---- B fragments: 16 contiguous bf16 (32B) per lane -> 2x b128 loads
        BF16x16 B0, B1;
        const __bf16* p0 = gb0 + jc + half * 16;
        const __bf16* p1 = gb1 + jc + half * 16;
        B0.q[0] = *(const v4u*)(p0);
        B0.q[1] = *(const v4u*)(p0 + 8);
        B1.q[0] = *(const v4u*)(p1);
        B1.q[1] = *(const v4u*)(p1 + 8);

        c0 = __builtin_amdgcn_wmma_f32_16x16x32_bf16(false, A.v, false, B0.v,
                                                     (short)0, c0, false, false);
        c1 = __builtin_amdgcn_wmma_f32_16x16x32_bf16(false, A.v, false, B1.v,
                                                     (short)0, c1, false, false);
    }

    // ---- epilogue: C layout: lane -> (M = r + 8*half, N = lane%16)
#pragma unroll
    for (int r = 0; r < 8; ++r) {
        const int M = r + half * 8;
        float* orow = out + (size_t)(i0 + M) * OUT_COLS + hh * N_HID;
        orow[rloc]      = c0[r];
        orow[16 + rloc] = c1[r];
    }
}

// ---------------------------------------------------------------------------
extern "C" void kernel_launch(void* const* d_in, const int* in_sizes, int n_in,
                              void* d_out, int out_size, void* d_ws, size_t ws_size,
                              hipStream_t stream)
{
    const float* h   = (const float*)d_in[0];
    const int*   adj = (const int*)d_in[1];
    const float* W   = (const float*)d_in[2];
    const float* a   = (const float*)d_in[3];
    float* out = (float*)d_out;

    char* ws = (char*)d_ws;
    __bf16* gT  = (__bf16*)(ws);                                 // 1 MB (256 x 2048 bf16)
    float*  s_l = (float*)(ws + 131072);                         // 64 KB
    float*  s_rT= (float*)(ws + 131072 + 65536);                 // 64 KB
    float*  m   = (float*)(ws + 131072 + 2 * 65536);             // 64 KB
    float*  iz  = (float*)(ws + 131072 + 3 * 65536);             // 64 KB

    k_gemm_g<<<256, 256, 0, stream>>>(h, W, gT);                 // 2048 waves
    k_scores<<<64, 256, 0, stream>>>(gT, a, s_l, s_rT);
    k_stats<<<2048, 256, 0, stream>>>(adj, s_l, s_rT, m, iz);
    // dynamic LDS: 4 KB adj double-buffer + 64 KB s_rT cache
    k_attn<<<128, 256, 4096 + 65536, stream>>>(adj, gT, s_l, s_rT, m, iz, out);
}
